// AeroModel_25159918420429
// MI455X (gfx1250) — matrix operand order repaired and checked by reading, not compiled, not run on hardware
//
#include <hip/hip_runtime.h>

typedef __attribute__((ext_vector_type(16))) __bf16   v16bf;
typedef __attribute__((ext_vector_type(8)))  __bf16   v8bf;
typedef __attribute__((ext_vector_type(8)))  float    v8f;
typedef __attribute__((ext_vector_type(4)))  float    v4f;
typedef __attribute__((ext_vector_type(4)))  unsigned v4u;

#define EPSV  1e-8f
#define SLOPE 0.01f

// d_ws layout:
//   floats [0..127]   : W1p[32][4]  = {W1[n][0..2], b1[n]}
//   floats [128..255] : Wd2t[32][4] = {Wd2[0][n], Wd2[1][n], Wd2[2][n], 0}
//   floats [256..287] : b2[32]
//   floats [288..319] : bd1[32]
//   floats [320..323] : bd2 (padded)
//   floats [324..327] : bias (padded)
//   byte 1536..5631   : bf16 B-operands, [4 tiles][32 lanes][16 bf16]
//                       tiles: {W2 ct0, W2 ct1, Wd1 ct0, Wd1 ct1}
#define WS_NCONST 328
#define WS_BF_OFF 384   // float index where bf16 block starts (byte 1536, 32B aligned)

__device__ __forceinline__ unsigned pk_bf2(float a, float b) {
    union { __bf16 h[2]; unsigned u; } x;
    x.h[0] = (__bf16)a; x.h[1] = (__bf16)b;
    return x.u;
}

// ---------------- one-block prep: pack consts + bf16 B-layout weights ----------------
__global__ __launch_bounds__(128) void aero_prep(
    const float* __restrict__ W1,  const float* __restrict__ b1,
    const float* __restrict__ W2,  const float* __restrict__ b2,
    const float* __restrict__ Wd1, const float* __restrict__ bd1,
    const float* __restrict__ Wd2, const float* __restrict__ bd2,
    const float* __restrict__ bias, float* __restrict__ wsf)
{
    const int t = threadIdx.x;
    if (t < 32) {
        wsf[t*4+0] = W1[t*3+0]; wsf[t*4+1] = W1[t*3+1];
        wsf[t*4+2] = W1[t*3+2]; wsf[t*4+3] = b1[t];
        wsf[128+t*4+0] = Wd2[0*32+t]; wsf[128+t*4+1] = Wd2[1*32+t];
        wsf[128+t*4+2] = Wd2[2*32+t]; wsf[128+t*4+3] = 0.f;
        wsf[256+t] = b2[t];
        wsf[288+t] = bd1[t];
        if (t < 4) {
            wsf[320+t] = (t < 3) ? bd2[t]  : 0.f;
            wsf[324+t] = (t < 3) ? bias[t] : 0.f;
        }
    }
    // bf16 B operands: B[k][n] = W[n][k]; per lane n = lmod+16*ct, k = 8*lhalf + (e&7) + 16*(e>>3)
    __bf16* wsb = (__bf16*)(wsf + WS_BF_OFF);
    const int tile = t >> 5;           // 0..3
    const int lane = t & 31;
    const int ct   = tile & 1;
    const float* W = (tile < 2) ? W2 : Wd1;
    const int n = (lane & 15) + 16*ct;
    const float* src = W + n*32 + 8*(lane >> 4);
    __bf16* dst = wsb + (tile*32 + lane)*16;
    #pragma unroll
    for (int e = 0; e < 16; ++e) dst[e] = (__bf16)src[(e & 7) + 16*(e >> 3)];
}

// ---------------- hot kernel ----------------
__global__ __launch_bounds__(256) void aero_mlp_kernel(
    const float* __restrict__ v, const float* __restrict__ w,
    const float* __restrict__ wsf, float* __restrict__ out)
{
    // Per-wave staging (wave-private -> DS in-order within a wave, no barriers needed)
    __shared__ __align__(16) __bf16 act[8][32][32];   // 16 KB: h1 then z, bf16 row-major
    __shared__ __align__(16) float  ybuf[8][32][32];  // 32 KB: layer-3 output, f32
    __shared__ __align__(16) float  cb[336];          // const block copy

    const int tid = threadIdx.x;
    #pragma unroll
    for (int i = tid; i < WS_NCONST; i += 256) cb[i] = wsf[i];
    __syncthreads();

    #define W1P(n,k)  cb[(n)*4+(k)]
    #define WD2T(n,k) cb[128+(n)*4+(k)]
    #define B2L(n)    cb[256+(n)]
    #define BD1L(n)   cb[288+(n)]
    #define BD2L(j)   cb[320+(j)]
    #define BIASL(j)  cb[324+(j)]

    const int wave  = tid >> 5;
    const int lane  = tid & 31;
    const int lhalf = lane >> 4;   // 0 or 1
    const int lmod  = lane & 15;

    // ---- B operands: direct loads of pre-laid-out bf16 (2x b128 each) ----
    const v16bf* bp = (const v16bf*)(wsf + WS_BF_OFF);   // each entry = 32 B
    const v16bf bW2_0  = bp[0*32 + lane];
    const v16bf bW2_1  = bp[1*32 + lane];
    const v16bf bWd1_0 = bp[2*32 + lane];
    const v16bf bWd1_1 = bp[3*32 + lane];

    // ---- Per-lane row: load, Gram-Schmidt ----
    const int row = blockIdx.x * 256 + wave*32 + lane;  // B divisible by 256: no tail
    const float vx = v[row*3+0], vy = v[row*3+1], vz = v[row*3+2];
    const float wx = w[row*3+0], wy = w[row*3+1], wz = w[row*3+2];

    const float nv  = sqrtf(vx*vx + vy*vy + vz*vz);
    const float iv  = 1.f / (nv + EPSV);
    const float v0 = vx*iv, v1 = vy*iv, v2 = vz*iv;
    const float proj = wx*v0 + wy*v1 + wz*v2;
    const float o0 = wx - proj*v0, o1 = wy - proj*v1, o2 = wz - proj*v2;
    const float nw  = sqrtf(o0*o0 + o1*o1 + o2*o2);
    const float iw  = 1.f / (nw + EPSV);
    const float w0 = o0*iw, w1 = o1*iw, w2 = o2*iw;
    const float u0 = v1*w2 - v2*w1;
    const float u1 = v2*w0 - v0*w2;
    const float u2 = v0*w1 - v1*w0;
    // feat = [v_on.v, v_on.w, w_on.w]
    const float f0 = v0*vx + v1*vy + v2*vz;
    const float f1 = v0*wx + v1*wy + v2*wz;
    const float f2 = w0*wx + w1*wy + w2*wz;

    // ---- Layer 1 (3->32) in VALU, pack h1 -> LDS bf16 row-major ----
    {
        float h1v[32];
        #pragma unroll
        for (int n = 0; n < 32; ++n) {
            float z = f0*W1P(n,0) + f1*W1P(n,1) + f2*W1P(n,2) + W1P(n,3);
            h1v[n] = (z >= 0.f) ? z : SLOPE*z;
        }
        v4u* dst = (v4u*)&act[wave][lane][0];
        #pragma unroll
        for (int g = 0; g < 4; ++g) {
            v4u q;
            q[0] = pk_bf2(h1v[g*8+0], h1v[g*8+1]);
            q[1] = pk_bf2(h1v[g*8+2], h1v[g*8+3]);
            q[2] = pk_bf2(h1v[g*8+4], h1v[g*8+5]);
            q[3] = pk_bf2(h1v[g*8+6], h1v[g*8+7]);
            dst[g] = q;
        }
    }

    // ---- Reload h1 in WMMA A layout (lane<16: K{0-7,16-23}; lane>=16: K{8-15,24-31}) ----
    const __bf16* abase = &act[wave][0][0];
    const int eoff = lhalf * 8;
    v8bf lo0 = *(const v8bf*)(abase + (lmod     )*32 + eoff);
    v8bf hi0 = *(const v8bf*)(abase + (lmod     )*32 + eoff + 16);
    v8bf lo1 = *(const v8bf*)(abase + (lmod + 16)*32 + eoff);
    v8bf hi1 = *(const v8bf*)(abase + (lmod + 16)*32 + eoff + 16);
    v16bf a0 = __builtin_shufflevector(lo0, hi0, 0,1,2,3,4,5,6,7,8,9,10,11,12,13,14,15);
    v16bf a1 = __builtin_shufflevector(lo1, hi1, 0,1,2,3,4,5,6,7,8,9,10,11,12,13,14,15);

    // ---- Layer 2: z = h1 @ W2^T + b2  (4x WMMA, C preloaded with bias splat) ----
    v8f acc[4];
    {
        const float bn0 = B2L(lmod), bn1 = B2L(lmod + 16);
        v8f c0 = {bn0,bn0,bn0,bn0,bn0,bn0,bn0,bn0};
        v8f c1 = {bn1,bn1,bn1,bn1,bn1,bn1,bn1,bn1};
        acc[0] = __builtin_amdgcn_wmma_f32_16x16x32_bf16(false, a0, false, bW2_0, (short)0, c0, false, false);
        acc[1] = __builtin_amdgcn_wmma_f32_16x16x32_bf16(false, a0, false, bW2_1, (short)0, c1, false, false);
        acc[2] = __builtin_amdgcn_wmma_f32_16x16x32_bf16(false, a1, false, bW2_0, (short)0, c0, false, false);
        acc[3] = __builtin_amdgcn_wmma_f32_16x16x32_bf16(false, a1, false, bW2_1, (short)0, c1, false, false);
    }

    // ---- Store z (bf16, row-major) from D layout ----
    #pragma unroll
    for (int rt = 0; rt < 2; ++rt)
        #pragma unroll
        for (int ct = 0; ct < 2; ++ct) {
            const int n = lmod + 16*ct;
            #pragma unroll
            for (int j = 0; j < 8; ++j) {
                const int m = rt*16 + j + 8*lhalf;
                act[wave][m][n] = (__bf16)acc[rt*2+ct][j];
            }
        }

    // ---- Reload z in A layout; elementwise h = leaky(z) * h1 in A domain ----
    v8bf zlo0 = *(const v8bf*)(abase + (lmod     )*32 + eoff);
    v8bf zhi0 = *(const v8bf*)(abase + (lmod     )*32 + eoff + 16);
    v8bf zlo1 = *(const v8bf*)(abase + (lmod + 16)*32 + eoff);
    v8bf zhi1 = *(const v8bf*)(abase + (lmod + 16)*32 + eoff + 16);
    v16bf za0 = __builtin_shufflevector(zlo0, zhi0, 0,1,2,3,4,5,6,7,8,9,10,11,12,13,14,15);
    v16bf za1 = __builtin_shufflevector(zlo1, zhi1, 0,1,2,3,4,5,6,7,8,9,10,11,12,13,14,15);

    v16bf ha0, ha1;
    #pragma unroll
    for (int e = 0; e < 16; ++e) {
        float z0 = (float)za0[e]; z0 = (z0 >= 0.f) ? z0 : SLOPE*z0;
        ha0[e] = (__bf16)(z0 * (float)a0[e]);
        float z1 = (float)za1[e]; z1 = (z1 >= 0.f) ? z1 : SLOPE*z1;
        ha1[e] = (__bf16)(z1 * (float)a1[e]);
    }

    // ---- Layer 3: zz = h @ Wd1^T + bd1  (4x WMMA) ----
    v8f acc2[4];
    {
        const float bn0 = BD1L(lmod), bn1 = BD1L(lmod + 16);
        v8f c0 = {bn0,bn0,bn0,bn0,bn0,bn0,bn0,bn0};
        v8f c1 = {bn1,bn1,bn1,bn1,bn1,bn1,bn1,bn1};
        acc2[0] = __builtin_amdgcn_wmma_f32_16x16x32_bf16(false, ha0, false, bWd1_0, (short)0, c0, false, false);
        acc2[1] = __builtin_amdgcn_wmma_f32_16x16x32_bf16(false, ha0, false, bWd1_1, (short)0, c1, false, false);
        acc2[2] = __builtin_amdgcn_wmma_f32_16x16x32_bf16(false, ha1, false, bWd1_0, (short)0, c0, false, false);
        acc2[3] = __builtin_amdgcn_wmma_f32_16x16x32_bf16(false, ha1, false, bWd1_1, (short)0, c1, false, false);
    }

    // ---- leaky + stage f32 row-major ----
    #pragma unroll
    for (int rt = 0; rt < 2; ++rt)
        #pragma unroll
        for (int ct = 0; ct < 2; ++ct) {
            const int n = lmod + 16*ct;
            #pragma unroll
            for (int j = 0; j < 8; ++j) {
                const int m = rt*16 + j + 8*lhalf;
                float t = acc2[rt*2+ct][j];
                ybuf[wave][m][n] = (t >= 0.f) ? t : SLOPE*t;
            }
        }

    // ---- Final layer (32->3) + rotation + bias, per lane on its own row ----
    float y0 = BD2L(0), y1 = BD2L(1), y2 = BD2L(2);
    const v4f* yr = (const v4f*)&ybuf[wave][lane][0];
    #pragma unroll
    for (int g = 0; g < 8; ++g) {
        v4f t = yr[g];
        #pragma unroll
        for (int k = 0; k < 4; ++k) {
            const int n = g*4 + k;
            y0 += t[k] * WD2T(n,0);
            y1 += t[k] * WD2T(n,1);
            y2 += t[k] * WD2T(n,2);
        }
    }
    // out = R @ y + bias, R columns = (v_on, w_on, u_on)
    out[row*3+0] = v0*y0 + w0*y1 + u0*y2 + BIASL(0);
    out[row*3+1] = v1*y0 + w1*y1 + u1*y2 + BIASL(1);
    out[row*3+2] = v2*y0 + w2*y1 + u2*y2 + BIASL(2);
}

extern "C" void kernel_launch(void* const* d_in, const int* in_sizes, int n_in,
                              void* d_out, int out_size, void* d_ws, size_t ws_size,
                              hipStream_t stream) {
    (void)n_in; (void)ws_size; (void)out_size;
    const float* v    = (const float*)d_in[0];
    const float* w    = (const float*)d_in[1];
    const float* W1   = (const float*)d_in[2];
    const float* b1   = (const float*)d_in[3];
    const float* W2   = (const float*)d_in[4];
    const float* b2   = (const float*)d_in[5];
    const float* Wd1  = (const float*)d_in[6];
    const float* bd1  = (const float*)d_in[7];
    const float* Wd2  = (const float*)d_in[8];
    const float* bd2  = (const float*)d_in[9];
    const float* bias = (const float*)d_in[10];
    float* out = (float*)d_out;
    float* wsf = (float*)d_ws;

    aero_prep<<<1, 128, 0, stream>>>(W1, b1, W2, b2, Wd1, bd1, Wd2, bd2, bias, wsf);

    const int rows   = in_sizes[0] / 3;   // 2,097,152
    const int blocks = rows / 256;        // 8192 (rows divisible by 256)
    aero_mlp_kernel<<<blocks, 256, 0, stream>>>(v, w, wsf, out);
}
